// mRNN_807453851871
// MI455X (gfx1250) — matrix operand order repaired and checked by reading, not compiled, not run on hardware
//
#include <hip/hip_runtime.h>

// ---------------- problem constants ----------------
#define Tt 1000
#define Bb 64
#define Nn 1024
#define Ii 128
#define ALPHA 0.1f
#define HID_SCALE 0.004472135955f   // sqrt(2*0.1)*0.01
#define INP_SCALE 0.04472135955f    // sqrt(2/0.1)*0.01

#define NT 64          // Nn/16  output tile columns
#define BT 4           // Bb/16  output tile rows
#define KT_REC 32      // Nn/32  k-tiles for recurrent GEMM
#define KT_INP 4       // Ii/32  k-tiles for input GEMM
#define WAVES 4
#define THREADS (WAVES*32)
#define BLOCKS (BT*NT) // 256 tiles -> 256 persistent blocks

typedef __attribute__((ext_vector_type(16))) __bf16   bf16x16;
typedef __attribute__((ext_vector_type(8)))  float    f32x8;
typedef __attribute__((ext_vector_type(4)))  unsigned uint32x4;

union Frag {                 // 32 bytes: one WMMA 16-bit A or B fragment per lane
    uint32x4 u4[2];
    bf16x16  v;
};

__device__ __forceinline__ unsigned short f2bf(float f) {
    unsigned u = __float_as_uint(f);
    u += 0x7FFFu + ((u >> 16) & 1u);          // round-to-nearest-even
    return (unsigned short)(u >> 16);
}

// ---------------- workspace layout (bytes) ----------------
#define WS_WREC   0u                               // 32*64*256*4  = 2097152
#define WS_WINP   2097152u                         // 4*64*256*4   = 262144
#define WS_U      2359296u                         // T*B*I*2      = 16384000
#define WS_STATE  18743296u                        // B*N*4        = 262144
#define WS_R0     19005440u                        // B*N*2        = 131072
#define WS_R1     19136512u                        // B*N*2        = 131072
#define WS_CNT    19267584u                        // 4-byte barrier counter

// Pack relu(W)*sign, transposed, into WMMA B-fragment order:
// entry index = ((kt*NT + nt)*32 + lane)*8 + j
// lane<16 : n_loc=lane,    k_pair = (j<4) ? 2j : 16+2(j-4)
// lane>=16: n_loc=lane-16, k_pair = (j<4) ? 8+2j : 24+2(j-4)
__global__ void pack_w_kernel(const float* __restrict__ W, const float* __restrict__ sgn,
                              unsigned* __restrict__ out, int K, int total) {
    int idx = blockIdx.x * blockDim.x + threadIdx.x;
    if (idx >= total) return;
    int j    = idx & 7;
    int lane = (idx >> 3) & 31;
    int nt   = (idx >> 8) & 63;
    int kt   = idx >> 14;
    int n  = nt * 16 + (lane & 15);
    int k0 = kt * 32 + ((j < 4) ? 8 * (lane >> 4) + 2 * j : 16 + 8 * (lane >> 4) + 2 * (j - 4));
    float w0 = __builtin_nontemporal_load(&W[(size_t)n * K + k0]);
    float w1 = __builtin_nontemporal_load(&W[(size_t)n * K + k0 + 1]);
    w0 = (w0 > 0.f ? w0 : 0.f) * sgn[k0];
    w1 = (w1 > 0.f ? w1 : 0.f) * sgn[k0 + 1];
    out[idx] = (unsigned)f2bf(w0) | ((unsigned)f2bf(w1) << 16);
}

// u_bf16[t][b][i] = bf16( inp[b][t][i] + INP_SCALE * noise_inp[t][b][i] )
__global__ void pack_u_kernel(const float* __restrict__ inp, const float* __restrict__ ninp,
                              unsigned short* __restrict__ u) {
    int idx = blockIdx.x * blockDim.x + threadIdx.x;   // idx = (t*B + b)*I + i
    if (idx >= Tt * Bb * Ii) return;
    int i  = idx % Ii;
    int tb = idx / Ii;
    int b  = tb % Bb;
    int t  = tb / Bb;
    float a = __builtin_nontemporal_load(&inp[((size_t)b * Tt + t) * Ii + i]);
    float e = __builtin_nontemporal_load(&ninp[idx]);
    u[idx] = f2bf(a + INP_SCALE * e);
}

__global__ void init_state_kernel(const float* __restrict__ h0,
                                  float* __restrict__ state,
                                  unsigned short* __restrict__ r0) {
    int idx = blockIdx.x * blockDim.x + threadIdx.x;
    if (idx >= Bb * Nn) return;
    float x = h0[idx];
    state[idx] = x;
    r0[idx] = f2bf(x > 0.f ? x : 0.f);
}

// ---------------- persistent RNN kernel ----------------
__global__ void __launch_bounds__(THREADS, 1)
rnn_persistent_kernel(const unsigned* __restrict__ wrec,
                      const unsigned* __restrict__ winp,
                      const unsigned short* __restrict__ u_all,
                      const float* __restrict__ bias,
                      const float* __restrict__ noise_hid,
                      float* __restrict__ out,
                      float* __restrict__ state,
                      unsigned short* __restrict__ rbuf0,
                      unsigned short* __restrict__ rbuf1,
                      unsigned* __restrict__ barrier_cnt) {
    const int wave = threadIdx.x >> 5;
    const int lane = threadIdx.x & 31;
    const int bt   = blockIdx.x >> 6;    // 0..3   batch tile
    const int nt   = blockIdx.x & 63;    // 0..63  N tile
    const int hi   = lane >> 4;          // lane half
    const int nloc = lane & 15;

    // --- hoist all weight fragments into registers for the whole run ---
    Frag Bf[9];
#pragma unroll
    for (int s = 0; s < 8; ++s) {
        int kt = wave + 4 * s;           // this wave's 8 recurrent k-tiles
        const uint32x4* p = (const uint32x4*)(wrec + ((((size_t)kt * NT + nt) * 32 + lane) * 8));
        Bf[s].u4[0] = p[0];
        Bf[s].u4[1] = p[1];
    }
    {
        const uint32x4* p = (const uint32x4*)(winp + ((((size_t)wave * NT + nt) * 32 + lane) * 8));
        Bf[8].u4[0] = p[0];
        Bf[8].u4[1] = p[1];
    }
    const float biasn = bias[nt * 16 + nloc];

    __shared__ float red[WAVES * 256];

    for (int t = 0; t < Tt; ++t) {
        const unsigned short* rcur  = (t & 1) ? rbuf1 : rbuf0;
        unsigned short*       rnext = (t & 1) ? rbuf0 : rbuf1;

        f32x8 acc = {0.f, 0.f, 0.f, 0.f, 0.f, 0.f, 0.f, 0.f};

        // recurrent GEMM: A = relu(x_{t-1}) in bf16, row-major [B][N] (hot in L2)
        const unsigned short* arow = rcur + ((size_t)(bt * 16 + nloc)) * Nn + hi * 8;
#pragma unroll
        for (int s = 0; s < 8; ++s) {
            int kt = wave + 4 * s;
            Frag A;
            A.u4[0] = *(const uint32x4*)(arow + kt * 32);
            A.u4[1] = *(const uint32x4*)(arow + kt * 32 + 16);
            acc = __builtin_amdgcn_wmma_f32_16x16x32_bf16(
                false, A.v, false, Bf[s].v, (short)0, acc, false, false);
        }
        // input GEMM: A = u_noisy(t) in bf16, row-major [B][I]
        {
            const unsigned short* urow =
                u_all + ((size_t)t * Bb + bt * 16 + nloc) * Ii + wave * 32 + hi * 8;
            Frag A;
            A.u4[0] = *(const uint32x4*)(urow);
            A.u4[1] = *(const uint32x4*)(urow + 16);
            acc = __builtin_amdgcn_wmma_f32_16x16x32_bf16(
                false, A.v, false, Bf[8].v, (short)0, acc, false, false);
        }

        // --- cross-wave K reduction through LDS ---
#pragma unroll
        for (int m = 0; m < 8; ++m) red[wave * 256 + m * 32 + lane] = acc[m];
        __syncthreads();

        // --- epilogue: each wave finalizes 2 of the 8 C rows ---
#pragma unroll
        for (int mi = 0; mi < 2; ++mi) {
            int m = wave * 2 + mi;
            float v = red[m * 32 + lane] + red[256 + m * 32 + lane] +
                      red[512 + m * 32 + lane] + red[768 + m * 32 + lane];
            int b = bt * 16 + m + 8 * hi;            // C layout: row = vgpr + 8*(lane>>4)
            int n = nt * 16 + nloc;
            size_t sidx = (size_t)b * Nn + n;
            float xo = state[sidx];
            // streaming read-once tensor: non-temporal, keep L2 for weights/r/u
            float nh = __builtin_nontemporal_load(
                &noise_hid[((size_t)t * Bb + b) * Nn + n]);
            float xn = (1.0f - ALPHA) * xo + ALPHA * (v + biasn) + HID_SCALE * nh;
            // streaming write-once trajectory: non-temporal store
            __builtin_nontemporal_store(xn, &out[((size_t)b * Tt + t) * Nn + n]);
            state[sidx] = xn;
            rnext[sidx] = f2bf(xn > 0.f ? xn : 0.f);
        }

        // --- device-wide barrier: write(t) must be visible before read(t+1) ---
        __threadfence();                 // each writer releases its stores
        __syncthreads();
        if (threadIdx.x == 0) {
            __hip_atomic_fetch_add(barrier_cnt, 1u, __ATOMIC_ACQ_REL,
                                   __HIP_MEMORY_SCOPE_AGENT);
            unsigned target = (unsigned)(t + 1) * BLOCKS;
            while (__hip_atomic_load(barrier_cnt, __ATOMIC_ACQUIRE,
                                     __HIP_MEMORY_SCOPE_AGENT) < target) {
                __builtin_amdgcn_s_sleep(1);
            }
        }
        __syncthreads();
        __threadfence();
    }
}

// ---------------- host launcher ----------------
extern "C" void kernel_launch(void* const* d_in, const int* in_sizes, int n_in,
                              void* d_out, int out_size, void* d_ws, size_t ws_size,
                              hipStream_t stream) {
    (void)in_sizes; (void)n_in; (void)out_size; (void)ws_size;
    const float* inp       = (const float*)d_in[0];  // [B,T,I]
    const float* h0        = (const float*)d_in[1];  // [B,N]
    const float* W_rec     = (const float*)d_in[2];  // [N,N]
    const float* W_inp     = (const float*)d_in[3];  // [N,I]
    const float* bias      = (const float*)d_in[4];  // [N]
    const float* sign_rec  = (const float*)d_in[5];  // [N]
    const float* sign_inp  = (const float*)d_in[6];  // [I]
    const float* noise_hid = (const float*)d_in[7];  // [T,B,N]
    const float* noise_inp = (const float*)d_in[8];  // [T,B,I]
    float* out = (float*)d_out;

    char* ws = (char*)d_ws;
    unsigned*       wrec  = (unsigned*)(ws + WS_WREC);
    unsigned*       winp  = (unsigned*)(ws + WS_WINP);
    unsigned short* u_all = (unsigned short*)(ws + WS_U);
    float*          state = (float*)(ws + WS_STATE);
    unsigned short* r0    = (unsigned short*)(ws + WS_R0);
    unsigned short* r1    = (unsigned short*)(ws + WS_R1);
    unsigned*       cnt   = (unsigned*)(ws + WS_CNT);

    hipMemsetAsync(cnt, 0, sizeof(unsigned), stream);   // deterministic barrier reset

    // Pack effective weights into WMMA B-fragment order (bf16, relu*sign fused).
    {
        int total = KT_REC * NT * 32 * 8;   // 524288
        pack_w_kernel<<<total / 256, 256, 0, stream>>>(W_rec, sign_rec, wrec, Nn, total);
    }
    {
        int total = KT_INP * NT * 32 * 8;   // 65536
        pack_w_kernel<<<total / 256, 256, 0, stream>>>(W_inp, sign_inp, winp, Ii, total);
    }
    // Precompute noisy inputs for all timesteps (parallel over T, outside the scan).
    {
        int total = Tt * Bb * Ii;           // 8192000
        pack_u_kernel<<<(total + 255) / 256, 256, 0, stream>>>(inp, noise_inp, u_all);
    }
    // Initial state and r = relu(h0).
    init_state_kernel<<<(Bb * Nn) / 256, 256, 0, stream>>>(h0, state, r0);

    // One persistent kernel runs the full 1000-step recurrence.
    rnn_persistent_kernel<<<BLOCKS, THREADS, 0, stream>>>(
        wrec, winp, u_all, bias, noise_hid, out, state, r0, r1, cnt);
}
